// Encoder_48421461295301
// MI455X (gfx1250) — matrix operand order repaired and checked by reading, not compiled, not run on hardware
//
#include <hip/hip_runtime.h>
#include <hip/hip_bf16.h>
#include <math.h>

// ---------------- problem constants ----------------
#define NUM_LAYERS 4
#define DMODEL 512
#define DIN    1024
#define NST    16
#define DTR    32
#define DCONV  4
#define BATCH  2
#define SEQ    512
#define BL     (BATCH*SEQ)          // 1024 tokens
#define BLD    (BATCH*SEQ*DMODEL)   // 524288

typedef __attribute__((ext_vector_type(16))) _Float16 v16h;
typedef __attribute__((ext_vector_type(8)))  float    v8f;

// ---------------- f32 -> f16 weight conversion ----------------
__global__ void cvt_f32_f16_k(const float* __restrict__ src, _Float16* __restrict__ dst, int n) {
    int i = blockIdx.x * 256 + threadIdx.x;
    if (i < n) dst[i] = (_Float16)src[i];
}

// ---------------- fused residual add + RMSNorm -> f16 token-major ----------------
__global__ void add_rmsnorm_k(const float* __restrict__ hs, float* __restrict__ residual,
                              const float* __restrict__ w, _Float16* __restrict__ hn16,
                              int first) {
    int t = blockIdx.x;               // token 0..BL-1
    int tid = threadIdx.x;
    size_t base = (size_t)t * DMODEL;
    float v0 = hs[base + tid];
    float v1 = hs[base + 256 + tid];
    if (!first) { v0 += residual[base + tid]; v1 += residual[base + 256 + tid]; }
    residual[base + tid] = v0;
    residual[base + 256 + tid] = v1;

    float ss = v0 * v0 + v1 * v1;
    #pragma unroll
    for (int o = 16; o > 0; o >>= 1) ss += __shfl_xor(ss, o, 32);
    __shared__ float red[8];
    if ((tid & 31) == 0) red[tid >> 5] = ss;
    __syncthreads();
    float tot = red[0];
    #pragma unroll
    for (int i = 1; i < 8; i++) tot += red[i];
    float scale = rsqrtf(tot * (1.0f / DMODEL) + 1e-5f);
    hn16[base + tid]       = (_Float16)(v0 * scale * w[tid]);
    hn16[base + 256 + tid] = (_Float16)(v1 * scale * w[256 + tid]);
}

// ---------------- final RMSNorm (fp32 out) ----------------
__global__ void final_norm_k(const float* __restrict__ hs, const float* __restrict__ residual,
                             const float* __restrict__ w, float* __restrict__ out) {
    int t = blockIdx.x;
    int tid = threadIdx.x;
    size_t base = (size_t)t * DMODEL;
    float v0 = hs[base + tid] + residual[base + tid];
    float v1 = hs[base + 256 + tid] + residual[base + 256 + tid];
    float ss = v0 * v0 + v1 * v1;
    #pragma unroll
    for (int o = 16; o > 0; o >>= 1) ss += __shfl_xor(ss, o, 32);
    __shared__ float red[8];
    if ((tid & 31) == 0) red[tid >> 5] = ss;
    __syncthreads();
    float tot = red[0];
    #pragma unroll
    for (int i = 1; i < 8; i++) tot += red[i];
    float scale = rsqrtf(tot * (1.0f / DMODEL) + 1e-5f);
    out[base + tid]       = v0 * scale * w[tid];
    out[base + 256 + tid] = v1 * scale * w[256 + tid];
}

// -------- register-blocked WMMA GEMM: C[M][N] = W[M][K] * Xt[N][K]^T --------
// One wave per (MT*16)x(NT*16) macro-tile: MT A-frags + NT B-frags per K-step
// feed MT*NT WMMAs (operand reuse in registers).
// MODE 0: plain row-major store          (in_proj -> xz[e][bl])
// MODE 1: + f16 transpose of rows < DTR  (x_proj  -> xdbl[r][bl], dt16[bl][r])
// MODE 2: softplus(acc + bias[m])        (dt_proj -> delta[d][bl])
// MODE 3: transposed store C[n*M + m]    (out_proj -> hs[bl][dm])
template <int MT, int NT, int MODE>
__global__ void wmma_gemm_k(const _Float16* __restrict__ W, const _Float16* __restrict__ Xt,
                            float* __restrict__ C, int M, int N, int K,
                            const float* __restrict__ bias, _Float16* __restrict__ aux) {
    int lane = threadIdx.x & 31;
    int lo = lane & 15;
    int hi = lane >> 4;                       // 0/1
    int m0 = blockIdx.x * (MT * 16);
    int n0 = blockIdx.y * (NT * 16);

    const _Float16* pa = W  + (size_t)(m0 + lo) * K + hi * 8;   // A row m, K halves per spec
    const _Float16* pb = Xt + (size_t)(n0 + lo) * K + hi * 8;   // B col n, K halves per spec

    v8f acc[MT][NT] = {};
    for (int k0 = 0; k0 < K; k0 += 32) {
        v16h a[MT], b[NT];
        #pragma unroll
        for (int i = 0; i < MT; i++) {
            const _Float16* p = pa + (size_t)i * 16 * K + k0;
            *(uint4*)&a[i]       = *(const uint4*)p;            // K  agrp*8   .. +7
            *((uint4*)&a[i] + 1) = *(const uint4*)(p + 16);     // K 16+agrp*8 .. +7
        }
        #pragma unroll
        for (int j = 0; j < NT; j++) {
            const _Float16* p = pb + (size_t)j * 16 * K + k0;
            *(uint4*)&b[j]       = *(const uint4*)p;            // K  bgrp*8    .. +7
            *((uint4*)&b[j] + 1) = *(const uint4*)(p + 16);     // K 16+bgrp*8 .. +7
        }
        #pragma unroll
        for (int i = 0; i < MT; i++)
            #pragma unroll
            for (int j = 0; j < NT; j++)
                acc[i][j] = __builtin_amdgcn_wmma_f32_16x16x32_f16(
                    false, a[i], false, b[j], (short)0, acc[i][j], false, false);
    }

    #pragma unroll
    for (int i = 0; i < MT; i++) {
        int mb = m0 + i * 16 + hi * 8;        // D layout: lane = n + (m>=8)*16, vgpr v -> m
        #pragma unroll
        for (int j = 0; j < NT; j++) {
            int ng = n0 + j * 16 + lo;
            #pragma unroll
            for (int v = 0; v < 8; v++) {
                int m = mb + v;
                float val = acc[i][j][v];
                if constexpr (MODE == 2) {
                    float xb = val + bias[m];
                    val = fmaxf(xb, 0.0f) + log1pf(expf(-fabsf(xb)));   // stable softplus
                }
                if constexpr (MODE == 3) {
                    C[(size_t)ng * M + m] = val;
                } else {
                    C[(size_t)m * N + ng] = val;
                    if constexpr (MODE == 1) {
                        if (m < DTR) aux[(size_t)ng * DTR + m] = (_Float16)val;
                    }
                }
            }
        }
    }
}

// ---------------- depthwise causal conv (dir-aware) + SiLU ----------------
__global__ void conv_silu_k(const float* __restrict__ xi, const float* __restrict__ cw,
                            const float* __restrict__ cb, int dir,
                            float* __restrict__ xc, _Float16* __restrict__ xcT) {
    int idx = blockIdx.x * 256 + threadIdx.x;        // DIN*BL
    int d  = idx >> 10;                              // / BL
    int bl = idx & (BL - 1);
    int b  = bl >> 9;                                // / SEQ
    int l  = bl & (SEQ - 1);
    const float* row = xi + (size_t)d * BL + b * SEQ;
    float acc = cb[d];
    #pragma unroll
    for (int j = 0; j < DCONV; j++) {
        int ls = l - (DCONV - 1) + j;
        if (ls >= 0) {
            int li = dir ? (SEQ - 1 - ls) : ls;
            acc = fmaf(cw[d * DCONV + j], row[li], acc);
        }
    }
    float s = acc / (1.0f + expf(-acc));             // SiLU
    xc[idx] = s;
    xcT[(size_t)bl * DIN + d] = (_Float16)s;
}

// ---------------- selective scan: one thread per (b,d), B/C staged in LDS ----
__global__ void scan_k(const float* __restrict__ u,     // xconv [DIN][BL]
                       const float* __restrict__ delta, // [DIN][BL]
                       const float* __restrict__ xdbl,  // [64][BL]: B rows 32.., C rows 48..
                       const float* __restrict__ Alog,  // [DIN][NST]
                       const float* __restrict__ Dskip, // [DIN]
                       const float* __restrict__ z,     // zi [DIN][BL]
                       int dir,
                       float* __restrict__ y) {         // [DIN][BL]
    int p = blockIdx.x * 256 + threadIdx.x;              // block = one b (DIN % 256 == 0)
    int b = p >> 10;
    int d = p & (DIN - 1);
    float A[NST], h[NST];
    #pragma unroll
    for (int n = 0; n < NST; n++) { A[n] = -expf(Alog[d * NST + n]); h[n] = 0.0f; }
    float Dv = Dskip[d];
    __shared__ float Bs[NST], Cs[NST];
    const float* Brow = xdbl + (size_t)DTR * BL;                  // rows 32..47
    const float* Crow = xdbl + (size_t)(DTR + NST) * BL;          // rows 48..63
    int base = b * SEQ;
    const size_t drow = (size_t)d * BL;
    for (int l = 0; l < SEQ; l++) {
        int bl = base + l;
        __syncthreads();
        if (threadIdx.x < NST)            Bs[threadIdx.x]       = Brow[threadIdx.x * BL + bl];
        else if (threadIdx.x < 2 * NST)   Cs[threadIdx.x - NST] = Crow[(threadIdx.x - NST) * BL + bl];
        __syncthreads();
        float dt = delta[drow + bl];
        float uv = u[drow + bl];
        if (l + 1 < SEQ) {                                        // gfx1250 global_prefetch
            __builtin_prefetch(&delta[drow + bl + 1], 0, 0);
            __builtin_prefetch(&u[drow + bl + 1], 0, 0);
        }
        float du = dt * uv;
        float yv = 0.0f;
        #pragma unroll
        for (int n = 0; n < NST; n++) {
            h[n] = fmaf(h[n], expf(dt * A[n]), du * Bs[n]);
            yv = fmaf(h[n], Cs[n], yv);
        }
        yv = fmaf(Dv, uv, yv);                                    // + D skip
        int zbl = base + (dir ? (SEQ - 1 - l) : l);
        float zv = z[drow + zbl];
        yv *= zv / (1.0f + expf(-zv));                            // * SiLU(z)
        y[drow + bl] = yv;
    }
}

// ---------------- combine fwd + reversed-bwd -> f16 token-major ----------------
__global__ void combine_k(const float* __restrict__ yf, const float* __restrict__ yr,
                          _Float16* __restrict__ y16) {
    int idx = blockIdx.x * 256 + threadIdx.x;        // DIN*BL
    int e  = idx >> 10;
    int bl = idx & (BL - 1);
    int b  = bl >> 9;
    int l  = bl & (SEQ - 1);
    float v = yf[idx] + yr[(size_t)e * BL + b * SEQ + (SEQ - 1 - l)];
    y16[(size_t)bl * DIN + e] = (_Float16)v;
}

// ---------------- host launch ----------------
static inline char* carve(char*& p, size_t bytes) {
    char* r = p;
    p += (bytes + 255) & ~(size_t)255;
    return r;
}

extern "C" void kernel_launch(void* const* d_in, const int* in_sizes, int n_in,
                              void* d_out, int out_size, void* d_ws, size_t ws_size,
                              hipStream_t stream) {
    const float* x        = (const float*)d_in[0];
    const float* norm_w   = (const float*)d_in[1];
    const float* in_proj  = (const float*)d_in[2];
    const float* convw_f  = (const float*)d_in[3];
    const float* convb_f  = (const float*)d_in[4];
    const float* xpw_f    = (const float*)d_in[5];
    const float* dtw_f    = (const float*)d_in[6];
    const float* dtb_f    = (const float*)d_in[7];
    const float* Alog_f   = (const float*)d_in[8];
    const float* Dskip_f  = (const float*)d_in[9];
    const float* convw_r  = (const float*)d_in[10];
    const float* convb_r  = (const float*)d_in[11];
    const float* xpw_r    = (const float*)d_in[12];
    const float* dtw_r    = (const float*)d_in[13];
    const float* dtb_r    = (const float*)d_in[14];
    const float* Alog_r   = (const float*)d_in[15];
    const float* Dskip_r  = (const float*)d_in[16];
    const float* out_proj = (const float*)d_in[17];
    const float* normf_w  = (const float*)d_in[18];
    float* out_f = (float*)d_out;

    char* wp = (char*)d_ws;
    float*    residual = (float*)carve(wp, sizeof(float) * BLD);
    _Float16* hn16     = (_Float16*)carve(wp, sizeof(_Float16) * (size_t)BL * DMODEL);
    float*    xz       = (float*)carve(wp, sizeof(float) * (size_t)2 * DIN * BL);
    float*    xconv[2]; _Float16* xcT[2]; float* xdbl[2]; _Float16* dt16[2];
    float*    delta[2]; float* ybr[2];
    for (int s = 0; s < 2; s++) {
        xconv[s] = (float*)carve(wp, sizeof(float) * (size_t)DIN * BL);
        xcT[s]   = (_Float16*)carve(wp, sizeof(_Float16) * (size_t)BL * DIN);
        xdbl[s]  = (float*)carve(wp, sizeof(float) * (size_t)64 * BL);
        dt16[s]  = (_Float16*)carve(wp, sizeof(_Float16) * (size_t)BL * DTR);
        delta[s] = (float*)carve(wp, sizeof(float) * (size_t)DIN * BL);
        ybr[s]   = (float*)carve(wp, sizeof(float) * (size_t)DIN * BL);
    }
    _Float16* y16 = (_Float16*)carve(wp, sizeof(_Float16) * (size_t)BL * DIN);
    // f16 weight mirrors
    const size_t n_inproj = (size_t)NUM_LAYERS * 2 * DIN * DMODEL;
    const size_t n_xpw    = (size_t)NUM_LAYERS * (DTR + 2 * NST) * DIN;
    const size_t n_dtw    = (size_t)NUM_LAYERS * DIN * DTR;
    const size_t n_outp   = (size_t)NUM_LAYERS * DMODEL * DIN;
    _Float16* inproj16 = (_Float16*)carve(wp, sizeof(_Float16) * n_inproj);
    _Float16* xpw16[2] = { (_Float16*)carve(wp, sizeof(_Float16) * n_xpw),
                           (_Float16*)carve(wp, sizeof(_Float16) * n_xpw) };
    _Float16* dtw16[2] = { (_Float16*)carve(wp, sizeof(_Float16) * n_dtw),
                           (_Float16*)carve(wp, sizeof(_Float16) * n_dtw) };
    _Float16* outp16   = (_Float16*)carve(wp, sizeof(_Float16) * n_outp);

    auto cvt = [&](const float* s, _Float16* dst, size_t n) {
        cvt_f32_f16_k<<<dim3((unsigned)((n + 255) / 256)), dim3(256), 0, stream>>>(s, dst, (int)n);
    };
    cvt(in_proj, inproj16, n_inproj);
    cvt(xpw_f, xpw16[0], n_xpw);  cvt(xpw_r, xpw16[1], n_xpw);
    cvt(dtw_f, dtw16[0], n_dtw);  cvt(dtw_r, dtw16[1], n_dtw);
    cvt(out_proj, outp16, n_outp);

    const float* convw[2] = { convw_f, convw_r };
    const float* convb[2] = { convb_f, convb_r };
    const float* dtb[2]   = { dtb_f, dtb_r };
    const float* Alog[2]  = { Alog_f, Alog_r };
    const float* Dsk[2]   = { Dskip_f, Dskip_r };

    for (int i = 0; i < NUM_LAYERS; i++) {
        const float* hs_in = (i == 0) ? x : (out_f + (size_t)i * BLD);   // feats[i-1]
        add_rmsnorm_k<<<dim3(BL), dim3(256), 0, stream>>>(
            hs_in, residual, norm_w + (size_t)i * DMODEL, hn16, (i == 0) ? 1 : 0);

        // in_proj: xz[2048][BL] = W(2048x512) * hn^T   (64x32 macro-tiles)
        wmma_gemm_k<4, 2, 0><<<dim3(2 * DIN / 64, BL / 32), dim3(32), 0, stream>>>(
            inproj16 + (size_t)i * 2 * DIN * DMODEL, hn16, xz,
            2 * DIN, BL, DMODEL, nullptr, nullptr);

        for (int s = 0; s < 2; s++) {
            conv_silu_k<<<dim3(DIN * BL / 256), dim3(256), 0, stream>>>(
                xz, convw[s] + (size_t)i * DIN * DCONV, convb[s] + (size_t)i * DIN,
                s, xconv[s], xcT[s]);
            // x_proj: xdbl[64][BL], also dt16[BL][32]  (M=64 -> no M-blocking)
            wmma_gemm_k<1, 1, 1><<<dim3(64 / 16, BL / 16), dim3(32), 0, stream>>>(
                xpw16[s] + (size_t)i * (DTR + 2 * NST) * DIN, xcT[s], xdbl[s],
                64, BL, DIN, nullptr, dt16[s]);
            // dt_proj + softplus: delta[DIN][BL]  (32x32 macro-tiles, single K-step)
            wmma_gemm_k<2, 2, 2><<<dim3(DIN / 32, BL / 32), dim3(32), 0, stream>>>(
                dtw16[s] + (size_t)i * DIN * DTR, dt16[s], delta[s],
                DIN, BL, DTR, dtb[s] + (size_t)i * DIN, nullptr);
            // selective scan (+ D skip, * SiLU(z))
            scan_k<<<dim3(BATCH * DIN / 256), dim3(256), 0, stream>>>(
                xconv[s], delta[s], xdbl[s],
                Alog[s] + (size_t)i * DIN * NST, Dsk[s] + (size_t)i * DIN,
                xz + (size_t)DIN * BL /* zi */, s, ybr[s]);
        }

        combine_k<<<dim3(DIN * BL / 256), dim3(256), 0, stream>>>(ybr[0], ybr[1], y16);

        // out_proj -> hs stored directly into feats[i] (B,L,D)  (32x32 macro-tiles)
        wmma_gemm_k<2, 2, 3><<<dim3(DMODEL / 32, BL / 32), dim3(32), 0, stream>>>(
            outp16 + (size_t)i * DMODEL * DIN, y16, out_f + (size_t)(1 + i) * BLD,
            DMODEL, BL, DIN, nullptr, nullptr);
    }

    final_norm_k<<<dim3(BL), dim3(256), 0, stream>>>(
        out_f + (size_t)NUM_LAYERS * BLD, residual, normf_w, out_f);
}
